// C_SS2D_25718264169245
// MI455X (gfx1250) — matrix operand order repaired and checked by reading, not compiled, not run on hardware
//
#include <hip/hip_runtime.h>
#include <hip/hip_bf16.h>

typedef __attribute__((ext_vector_type(16))) _Float16 v16h;
typedef __attribute__((ext_vector_type(8)))  float    v8f;

// Problem constants (match setup_inputs()).
#define LC    4096   // H*W
#define HH    64
#define WW    64
#define CC    96     // channels per half
#define DSN   8      // d_state
#define DTR   12     // dt_rank
#define PROJ  28     // dt_rank + 2*d_state
#define KDIR  4
#define NHALF 2
#define NB    2

// Workspace layout (floats)
#define XDBL_ELEMS  (NHALF*NB*KDIR*PROJ*LC)   // 1,835,008
#define DELTA_ELEMS (NHALF*NB*KDIR*CC*LC)     // 6,291,456
#define YK_ELEMS    (NHALF*NB*KDIR*CC*LC)

// scan index t (direction k) -> linear position inside the (H,W) image
__device__ __forceinline__ int scan_pos(int k, int t) {
    int u = (k & 2) ? (LC - 1 - t) : t;          // k=2,3 scan the flipped sequence
    return (k & 1) ? ((u & (HH - 1)) * WW + (u >> 6))  // k=1,3: W/H-transposed order
                   : u;                                 // k=0,2: row-major
}

__device__ __forceinline__ float softplus_f(float x) {
    return (x > 20.f) ? x : log1pf(__expf(x));
}

// ---------------------------------------------------------------------------
// Kernel 1: x_dbl[half][b][k][28][L] = Wproj_k(28x96) @ xs_k(96xL)  (WMMA f16)
// One wave per 16-wide N tile; it produces both 16-row M tiles, reusing B regs.
// ---------------------------------------------------------------------------
__global__ void ss2d_proj1(const float* __restrict__ Fin,
                           const float* __restrict__ Wproj,   // (4,28,96)
                           float* __restrict__ x_dbl) {
    const int wave = (blockIdx.x * blockDim.x + threadIdx.x) >> 5;
    const int lane = threadIdx.x & 31;
    const int nt   = wave & 255;
    const int k    = (wave >> 8) & 3;
    const int b    = (wave >> 10) & 1;
    const int half = (wave >> 11) & 1;
    const int mrow = lane & 15;        // N column for B/D, M row for A
    const int side = lane >> 4;

    const float* finb = Fin + (size_t)(b * 2 * CC + half * CC) * LC;
    const int l = nt * 16 + mrow;
    const int pos = scan_pos(k, l);

    // B operand: xs_k tile (32 x 16) per k-step; element e -> K, lane -> N.
    v16h Bt[3];
#pragma unroll
    for (int kt = 0; kt < 3; ++kt) {
#pragma unroll
        for (int e = 0; e < 16; ++e) {
            int ci = kt * 32 + ((e < 8) ? (e + side * 8) : (16 + (e - 8) + side * 8));
            Bt[kt][e] = (_Float16)finb[(size_t)ci * LC + pos];
        }
    }

    const float* Wk = Wproj + k * PROJ * CC;
    float* xd = x_dbl + (size_t)(((half * NB + b) * KDIR + k) * PROJ) * LC;

#pragma unroll
    for (int mt = 0; mt < 2; ++mt) {
        v8f acc = {};
        const int m = mt * 16 + mrow;
#pragma unroll
        for (int kt = 0; kt < 3; ++kt) {
            v16h At;
#pragma unroll
            for (int e = 0; e < 16; ++e) {
                int kloc = (e < 8) ? (e + side * 8) : (16 + (e - 8) + side * 8);
                int ci = kt * 32 + kloc;
                At[e] = (m < PROJ) ? (_Float16)Wk[m * CC + ci] : (_Float16)0.f;
            }
            acc = __builtin_amdgcn_wmma_f32_16x16x32_f16(
                false, At, false, Bt[kt], (short)0, acc, false, false);
        }
#pragma unroll
        for (int r = 0; r < 8; ++r) {
            int row = mt * 16 + r + side * 8;
            if (row < PROJ)
                xd[(size_t)row * LC + nt * 16 + mrow] = acc[r];
        }
    }
}

// ---------------------------------------------------------------------------
// Kernel 2: delta[half][b][k][96][L] = softplus(dtW_k(96x12) @ dts(12xL) + bias)
// K padded 12->32 with zeros; 6 M tiles per wave, B regs reused.
// ---------------------------------------------------------------------------
__global__ void ss2d_proj2(const float* __restrict__ x_dbl,
                           const float* __restrict__ dtW,     // (4,96,12)
                           const float* __restrict__ dtBias,  // (4,96)
                           float* __restrict__ delta) {
    const int wave = (blockIdx.x * blockDim.x + threadIdx.x) >> 5;
    const int lane = threadIdx.x & 31;
    const int nt   = wave & 255;
    const int k    = (wave >> 8) & 3;
    const int b    = (wave >> 10) & 1;
    const int half = (wave >> 11) & 1;
    const int mrow = lane & 15;
    const int side = lane >> 4;

    const float* xd = x_dbl + (size_t)(((half * NB + b) * KDIR + k) * PROJ) * LC;
    float* dl = delta + (size_t)(((half * NB + b) * KDIR + k) * CC) * LC;
    const float* Wk = dtW + k * CC * DTR;
    const float* bk = dtBias + k * CC;
    const int col = nt * 16 + mrow;

    v16h Bt;   // dts tile (rows 0..11 of x_dbl), zero-padded K 12..31
#pragma unroll
    for (int e = 0; e < 16; ++e) {
        int kk = (e < 8) ? (e + side * 8) : (16 + (e - 8) + side * 8);
        Bt[e] = (kk < DTR) ? (_Float16)xd[(size_t)kk * LC + col] : (_Float16)0.f;
    }

#pragma unroll
    for (int mt = 0; mt < 6; ++mt) {
        const int m = mt * 16 + mrow;
        v16h At;
#pragma unroll
        for (int e = 0; e < 16; ++e) {
            int kloc = (e < 8) ? (e + side * 8) : (16 + (e - 8) + side * 8);
            At[e] = (kloc < DTR) ? (_Float16)Wk[m * DTR + kloc] : (_Float16)0.f;
        }
        v8f acc = {};
        acc = __builtin_amdgcn_wmma_f32_16x16x32_f16(
            false, At, false, Bt, (short)0, acc, false, false);
#pragma unroll
        for (int r = 0; r < 8; ++r) {
            int row = mt * 16 + r + side * 8;        // < 96 always
            dl[(size_t)row * LC + col] = softplus_f(acc[r] + bk[row]);
        }
    }
}

// ---------------------------------------------------------------------------
// Kernel 3: sequential scan. One block per (half,b,k); 768 threads, one per
// (d,n) state element. 8-lane d-groups live inside a wave32 -> the MLP and the
// C-contraction use lane shuffles only (no barriers inside the 4096-step loop).
// ---------------------------------------------------------------------------
__global__ void __launch_bounds__(768)
ss2d_scan(const float* __restrict__ Fin,
          const float* __restrict__ x_dbl,
          const float* __restrict__ delta,
          const float* __restrict__ A_logs,  // (4,96,8)
          const float* __restrict__ Ds,      // (4,96)
          const float* __restrict__ w1, const float* __restrict__ b1,
          const float* __restrict__ w2, const float* __restrict__ b2,
          const float* __restrict__ w3, const float* __restrict__ b3,
          float* __restrict__ yk) {
    const int k    = blockIdx.x & 3;
    const int b    = (blockIdx.x >> 2) & 1;
    const int half = blockIdx.x >> 3;
    const int tid  = threadIdx.x;
    const int n    = tid & 7;           // state index
    const int d    = tid >> 3;          // channel
    const int lane = tid & 31;
    const int base = lane & ~7;         // first lane of this d-group

    // per-thread constants
    float w1r[8], w2r[8], w3r[8];
#pragma unroll
    for (int i = 0; i < 8; ++i) {
        w1r[i] = w1[n * 8 + i];
        w2r[i] = w2[n * 8 + i];
        w3r[i] = w3[n * 8 + i];
    }
    const float b1v = b1[n], b2v = b2[n], b3v = b3[n];
    const float Av  = -__expf(A_logs[(k * CC + d) * DSN + n]);
    const float Dsv = Ds[k * CC + d];

    const size_t hbk = (size_t)((half * NB + b) * KDIR + k);
    const float* dlt = delta + (hbk * CC + d) * LC;
    const float* xdb = x_dbl + hbk * PROJ * LC;
    const float* Bp  = xdb + (size_t)(DTR + n) * LC;
    const float* Cp  = xdb + (size_t)(DTR + DSN + n) * LC;
    const float* xrw = Fin + (size_t)(b * 2 * CC + half * CC + d) * LC;
    float* yrow      = yk + (hbk * CC + d) * LC;

    float h = 0.f;
    for (int t = 0; t < LC; ++t) {
        if ((t & 63) == 0 && t + 256 < LC) {           // CDNA5 global_prefetch_b8
            __builtin_prefetch(dlt + t + 256, 0, 1);
            __builtin_prefetch(Bp + t + 256, 0, 1);
            __builtin_prefetch(Cp + t + 256, 0, 1);
        }
        const float dv = dlt[t];
        const float Bv = Bp[t];
        const float Cv = Cp[t];
        const float xv = xrw[scan_pos(k, t)];

        // MLP(h) across the 8-lane d-group
        float t1 = b1v;
#pragma unroll
        for (int j = 0; j < 8; ++j) t1 += __shfl(h, base + j, 32) * w1r[j];
        t1 = fmaxf(t1, 0.f);
        float t2 = b2v;
#pragma unroll
        for (int j = 0; j < 8; ++j) t2 += __shfl(t1, base + j, 32) * w2r[j];
        t2 = fmaxf(t2, 0.f);
        float t3 = b3v;
#pragma unroll
        for (int j = 0; j < 8; ++j) t3 += __shfl(t2, base + j, 32) * w3r[j];

        const float dA = __expf(dv * Av);
        h = dA * h + t3 * (dv * Bv * xv);

        // y = sum_n h*C  (xor-reduction over the 8-lane group)
        float yv = h * Cv;
        yv += __shfl_xor(yv, 1, 32);
        yv += __shfl_xor(yv, 2, 32);
        yv += __shfl_xor(yv, 4, 32);
        if (n == 0) yrow[t] = yv + xv * Dsv;           // out = y + xs*Ds
    }
}

// ---------------------------------------------------------------------------
// Kernel 4: merge the 4 directions + LayerNorm over c=96, write final output.
// 128 threads (96 active) per (half,b,l).
// ---------------------------------------------------------------------------
__global__ void __launch_bounds__(128)
ss2d_merge_ln(const float* __restrict__ yk,
              const float* __restrict__ gamma,
              const float* __restrict__ beta,
              float* __restrict__ out) {
    const int l    = blockIdx.x & (LC - 1);
    const int b    = (blockIdx.x >> 12) & 1;
    const int half = blockIdx.x >> 13;
    const int d    = threadIdx.x;
    const bool act = (d < CC);

    const int h = l >> 6, w = l & (WW - 1);
    const int lT = w * HH + h;                 // position in transposed scan

    float v = 0.f;
    if (act) {
        const size_t hb = (size_t)(half * NB + b) * KDIR;
        v  = yk[((hb + 0) * CC + d) * LC + l];
        v += yk[((hb + 2) * CC + d) * LC + (LC - 1 - l)];
        v += yk[((hb + 1) * CC + d) * LC + lT];
        v += yk[((hb + 3) * CC + d) * LC + (LC - 1 - lT)];
    }

    __shared__ float ssum[128], ssq[128];
    ssum[threadIdx.x] = v;
    ssq[threadIdx.x]  = v * v;
    __syncthreads();
#pragma unroll
    for (int s = 64; s > 0; s >>= 1) {
        if (threadIdx.x < s) {
            ssum[threadIdx.x] += ssum[threadIdx.x + s];
            ssq[threadIdx.x]  += ssq[threadIdx.x + s];
        }
        __syncthreads();
    }
    const float mean = ssum[0] / (float)CC;
    const float var  = ssq[0] / (float)CC - mean * mean;
    if (act) {
        float r = (v - mean) * rsqrtf(var + 1e-5f) * gamma[d] + beta[d];
        out[((size_t)half * NB * LC + (size_t)b * LC + l) * CC + d] = r;
    }
}

// ---------------------------------------------------------------------------
extern "C" void kernel_launch(void* const* d_in, const int* in_sizes, int n_in,
                              void* d_out, int out_size, void* d_ws, size_t ws_size,
                              hipStream_t stream) {
    (void)in_sizes; (void)n_in; (void)out_size; (void)ws_size;
    const float* Fin    = (const float*)d_in[0];
    const float* xprojW = (const float*)d_in[1];
    const float* dtW    = (const float*)d_in[2];
    const float* dtB    = (const float*)d_in[3];
    const float* Alogs  = (const float*)d_in[4];
    const float* Ds     = (const float*)d_in[5];
    const float* w1     = (const float*)d_in[6];
    const float* b1     = (const float*)d_in[7];
    const float* w2     = (const float*)d_in[8];
    const float* b2     = (const float*)d_in[9];
    const float* w3     = (const float*)d_in[10];
    const float* b3     = (const float*)d_in[11];
    const float* gamma  = (const float*)d_in[12];
    const float* beta   = (const float*)d_in[13];
    float* out = (float*)d_out;

    float* wsf   = (float*)d_ws;
    float* x_dbl = wsf;                              // 1,835,008 f32
    float* delta = x_dbl + XDBL_ELEMS;               // 6,291,456 f32
    float* yk    = delta + DELTA_ELEMS;              // 6,291,456 f32

    // 4096 wave-jobs, 8 waves / 256-thread block -> 512 blocks
    ss2d_proj1<<<512, 256, 0, stream>>>(Fin, xprojW, x_dbl);
    ss2d_proj2<<<512, 256, 0, stream>>>(x_dbl, dtW, dtB, delta);
    ss2d_scan<<<NHALF * NB * KDIR, 768, 0, stream>>>(
        Fin, x_dbl, delta, Alogs, Ds, w1, b1, w2, b2, w3, b3, yk);
    ss2d_merge_ln<<<NHALF * NB * LC, 128, 0, stream>>>(yk, gamma, beta, out);
}
